// SPN_16673063043389
// MI455X (gfx1250) — matrix operand order repaired and checked by reading, not compiled, not run on hardware
//
#include <hip/hip_runtime.h>
#include <math.h>

// ---------------------------------------------------------------------------
// SPN segmentation net on gfx1250 (MI455X).
// Conv = im2col (f32->f16, K zero-padded to 32) + LDS-tiled WMMA GEMM:
//   block = 128 threads (4 waves), 128x64 output tile, K chunks of 32,
//   double-buffered LDS (one barrier per chunk), 8 x v_wmma_f32_16x16x32_f16
//   per wave per chunk (2 A-frags x 4 B-frags), f32 accumulate,
//   fused bias/ReLU/ELU/residual + NCHW scatter in the epilogue.
// ---------------------------------------------------------------------------

typedef __attribute__((ext_vector_type(16))) _Float16 v16h;
typedef __attribute__((ext_vector_type(8)))  _Float16 v8h;
typedef __attribute__((ext_vector_type(8)))  float    v8f;

#define ACT_NONE 0
#define ACT_RELU 1
#define ACT_ELU  2

#define BM 128          // block tile M (pixels)
#define BN 64           // block tile N (output channels)
#define LDS_STRIDE 40   // 32 K-halves + 8 pad (80B row stride: conflict spread)

static __device__ inline v16h combine16(v8h lo, v8h hi)
{
    return __builtin_shufflevector(lo, hi, 0, 1, 2, 3, 4, 5, 6, 7,
                                   8, 9, 10, 11, 12, 13, 14, 15);
}

// ---------------------------------------------------------------------------
// im2col, f32 -> f16, row-major [M][Kpad], kk = ci*K*K + kh*K + kw,
// zero fill for kk >= KK. grid = (M, ceil(Kpad/256)), block = 256.
// ---------------------------------------------------------------------------
__global__ void im2col_f16_kernel(const float* __restrict__ in, _Float16* __restrict__ col,
                                  int Cin, int Hin, int Win, int Hout, int Wout,
                                  int K, int stride, int pad, int KK, int Kpad)
{
    const int m  = blockIdx.x;
    const int kk = blockIdx.y * blockDim.x + threadIdx.x;
    if (kk >= Kpad) return;
    const int HWo = Hout * Wout;
    const int b  = m / HWo;
    const int r  = m - b * HWo;
    const int oy = r / Wout;
    const int ox = r - oy * Wout;
    float v = 0.0f;
    if (kk < KK) {
        const int KK2 = K * K;
        const int ci = kk / KK2;
        const int rr = kk - ci * KK2;
        const int kh = rr / K;
        const int kw = rr - kh * K;
        const int iy = oy * stride - pad + kh;
        const int ix = ox * stride - pad + kw;
        if (iy >= 0 && iy < Hin && ix >= 0 && ix < Win)
            v = in[(((size_t)b * Cin + ci) * Hin + iy) * Win + ix];
    }
    col[(size_t)m * Kpad + kk] = (_Float16)v;
}

// ---------------------------------------------------------------------------
// Weight transform: OIHW f32 -> [Npad][Kpad] f16, zero padded.
// ---------------------------------------------------------------------------
__global__ void wxform_f16_kernel(const float* __restrict__ w, _Float16* __restrict__ wf,
                                  int Cout, int KK, int Kpad, int total)
{
    const int i = blockIdx.x * blockDim.x + threadIdx.x;
    if (i >= total) return;
    const int n  = i / Kpad;
    const int kk = i - n * Kpad;
    float v = 0.0f;
    if (n < Cout && kk < KK) v = w[(size_t)n * KK + kk];
    wf[i] = (_Float16)v;
}

// ---------------------------------------------------------------------------
// Double-buffered LDS WMMA GEMM with fused conv epilogue.
//   A: [M][Kpad] f16 (im2col), B: [Npad][Kpad] f16 (weights)
// CDNA5 wave32 fragment layouts (ISA 7.12.2):
//   A (16x32): lane=half*16+m; K slots = {half*8+0..7, 16+half*8+0..7}
//   B (32x16): lane=half*16+n; K slots = {half*8+0..7, 16+half*8+0..7}
//   D: col = lane&15, row = r + 8*(lane>>4).
// Each wave owns 32 rows (2 A-frags), all 64 cols (4 B-frags) -> 8 acc.
// ---------------------------------------------------------------------------
__global__ void __launch_bounds__(128) gemm_wmma_kernel(
    const _Float16* __restrict__ A, const _Float16* __restrict__ Bw,
    const float* __restrict__ bias, const float* __restrict__ residual,
    float* __restrict__ out,
    int M, int Cout, int Kpad,
    int Hout, int Wout, int act)
{
    __shared__ _Float16 As[2 * BM * LDS_STRIDE];   // 20 KB
    __shared__ _Float16 Bs[2 * BN * LDS_STRIDE];   // 10 KB

    const int tid  = threadIdx.x;
    const int wave = tid >> 5;
    const int lane = tid & 31;
    const int half = lane >> 4;
    const int lidx = lane & 15;
    const int m0 = blockIdx.x * BM;
    const int n0 = blockIdx.y * BN;

    // staging: thread tid moves A row tid (64B = 4x float4) and
    //          B row tid>>1, segment (tid&1)*16 halves (32B = 2x float4)
    const _Float16* gA = A  + (size_t)(m0 + tid) * Kpad;
    const _Float16* gB = Bw + (size_t)(n0 + (tid >> 1)) * Kpad + (tid & 1) * 16;
    _Float16* const sA = &As[tid * LDS_STRIDE];
    _Float16* const sB = &Bs[(tid >> 1) * LDS_STRIDE + (tid & 1) * 16];

    const int nch = Kpad >> 5;

    auto stage = [&](int chunk, int buf) {
        const float4* pa = (const float4*)(gA + chunk * 32);
        const float4* pb = (const float4*)(gB + chunk * 32);
        if (chunk + 1 < nch) {
            __builtin_prefetch(gA + (chunk + 1) * 32, 0, 1);
            __builtin_prefetch(gB + (chunk + 1) * 32, 0, 1);
        }
        const float4 a0 = pa[0], a1 = pa[1], a2 = pa[2], a3 = pa[3];
        const float4 b0 = pb[0], b1 = pb[1];
        float4* dA = (float4*)(sA + buf * (BM * LDS_STRIDE));
        float4* dB = (float4*)(sB + buf * (BN * LDS_STRIDE));
        dA[0] = a0; dA[1] = a1; dA[2] = a2; dA[3] = a3;
        dB[0] = b0; dB[1] = b1;
    };

    v8f acc[8];
    #pragma unroll
    for (int i = 0; i < 8; ++i) acc[i] = (v8f){};

    stage(0, 0);
    __syncthreads();

    const int aro0 = (wave * 32 + lidx) * LDS_STRIDE;
    const int aro1 = (wave * 32 + 16 + lidx) * LDS_STRIDE;

    for (int i = 0; i < nch; ++i) {
        const int p = i & 1;
        if (i + 1 < nch) stage(i + 1, p ^ 1);   // overlap with this chunk's math

        const _Float16* Ab = As + p * (BM * LDS_STRIDE);
        const _Float16* Bb = Bs + p * (BN * LDS_STRIDE);

        const v16h a0 = combine16(*(const v8h*)(Ab + aro0 + half * 8),
                                  *(const v8h*)(Ab + aro0 + 16 + half * 8));
        const v16h a1 = combine16(*(const v8h*)(Ab + aro1 + half * 8),
                                  *(const v8h*)(Ab + aro1 + 16 + half * 8));
        #pragma unroll
        for (int j = 0; j < 4; ++j) {
            const _Float16* rB = Bb + (j * 16 + lidx) * LDS_STRIDE + half * 8;
            const v16h bf = combine16(*(const v8h*)(rB), *(const v8h*)(rB + 16));
            acc[j]     = __builtin_amdgcn_wmma_f32_16x16x32_f16(false, a0, false, bf, (short)0, acc[j],     false, false);
            acc[4 + j] = __builtin_amdgcn_wmma_f32_16x16x32_f16(false, a1, false, bf, (short)0, acc[4 + j], false, false);
        }
        __syncthreads();   // reads of buf p done; buf p^1 stores visible
    }

    // ---- fused epilogue: bias + act + residual, scatter to NCHW ----
    const int HWo = Hout * Wout;
    #pragma unroll
    for (int sub = 0; sub < 2; ++sub) {
        #pragma unroll
        for (int j = 0; j < 4; ++j) {
            const v8f a = acc[sub * 4 + j];
            const int n = n0 + j * 16 + lidx;
            if (n >= Cout) continue;
            const float bb = bias ? bias[n] : 0.0f;
            #pragma unroll
            for (int r = 0; r < 8; ++r) {
                const int m = m0 + wave * 32 + sub * 16 + r + 8 * half;
                if (m >= M) continue;
                const int b  = m / HWo;
                const int rr = m - b * HWo;
                const int oy = rr / Wout;
                const int ox = rr - oy * Wout;
                float v = a[r] + bb;
                if (act == ACT_RELU)     v = fmaxf(v, 0.0f);
                else if (act == ACT_ELU) v = (v > 0.0f) ? v : (__expf(v) - 1.0f);
                const size_t oidx = (((size_t)b * Cout + n) * Hout + oy) * Wout + ox;
                if (residual) v += residual[oidx];
                out[oidx] = v;
            }
        }
    }
}

// ---------------------------------------------------------------------------
// Max-pool 3x3 / stride 2 / pad 1 (relu_in folds relu before the pool).
// ---------------------------------------------------------------------------
__global__ void maxpool_kernel(const float* __restrict__ in, float* __restrict__ out,
                               int BC, int Hin, int Win, int Hout, int Wout, int relu_in)
{
    const int i = blockIdx.x * blockDim.x + threadIdx.x;
    const int total = BC * Hout * Wout;
    if (i >= total) return;
    const int ox = i % Wout;
    int t = i / Wout;
    const int oy = t % Hout;
    const int bc = t / Hout;
    const int y0 = oy * 2 - 1, x0 = ox * 2 - 1;
    float m = -INFINITY;
    for (int dy = 0; dy < 3; ++dy) {
        const int y = y0 + dy;
        if (y < 0 || y >= Hin) continue;
        for (int dx = 0; dx < 3; ++dx) {
            const int x = x0 + dx;
            if (x < 0 || x >= Win) continue;
            m = fmaxf(m, in[((size_t)bc * Hin + y) * Win + x]);
        }
    }
    if (relu_in) m = fmaxf(m, 0.0f);
    out[i] = m;
}

// ---------------------------------------------------------------------------
// Bilinear 2x upsample (half-pixel centers, edge clamp), optional residual.
// ---------------------------------------------------------------------------
__global__ void up2_kernel(const float* __restrict__ in, const float* __restrict__ res,
                           float* __restrict__ out, int BC, int Hin, int Win)
{
    const int Hout = Hin * 2, Wout = Win * 2;
    const int i = blockIdx.x * blockDim.x + threadIdx.x;
    const int total = BC * Hout * Wout;
    if (i >= total) return;
    const int ox = i % Wout;
    int t = i / Wout;
    const int oy = t % Hout;
    const int bc = t / Hout;
    const float sy = (oy + 0.5f) * 0.5f - 0.5f;
    const float sx = (ox + 0.5f) * 0.5f - 0.5f;
    const int y0 = (int)floorf(sy);
    const int x0 = (int)floorf(sx);
    const float fy = sy - (float)y0;
    const float fx = sx - (float)x0;
    const int y0c = min(max(y0, 0), Hin - 1);
    const int y1c = min(max(y0 + 1, 0), Hin - 1);
    const int x0c = min(max(x0, 0), Win - 1);
    const int x1c = min(max(x0 + 1, 0), Win - 1);
    const float* p = in + (size_t)bc * Hin * Win;
    float v = (1.0f - fy) * ((1.0f - fx) * p[y0c * Win + x0c] + fx * p[y0c * Win + x1c])
            +         fy  * ((1.0f - fx) * p[y1c * Win + x0c] + fx * p[y1c * Win + x1c]);
    if (res) v += res[i];
    out[i] = v;
}

// ---------------------------------------------------------------------------
// Gated spatial-propagation scan (one block per (b,c); LDS-resident column).
// ---------------------------------------------------------------------------
__global__ void __launch_bounds__(128) spn_scan_kernel(
    const float* __restrict__ X, const float* __restrict__ G,
    float* __restrict__ out,
    int C, int GC, int g1c, int g2c, int g3c,
    int H, int W, int horizontal, int reverse)
{
    __shared__ float hcol[130];
    const int bc = blockIdx.x;
    const int b = bc / C;
    const int c = bc - b * C;
    const int t = threadIdx.x;
    const int len   = horizontal ? H : W;
    const int steps = horizontal ? W : H;
    if (t == 0) { hcol[0] = 0.0f; hcol[len + 1] = 0.0f; }
    hcol[t + 1] = 0.0f;
    __syncthreads();

    const size_t xBase  = ((size_t)b * C + c) * H * W;
    const size_t g1Base = ((size_t)b * GC + g1c + c) * H * W;
    const size_t g2Base = ((size_t)b * GC + g2c + c) * H * W;
    const size_t g3Base = ((size_t)b * GC + g3c + c) * H * W;

    for (int s = 0; s < steps; ++s) {
        const int p = reverse ? (steps - 1 - s) : s;
        const int y = horizontal ? t : p;
        const int x = horizontal ? p : t;
        const size_t idx = (size_t)y * W + x;
        const float xt = X[xBase + idx];
        float g1 = G[g1Base + idx];
        float g2 = G[g2Base + idx];
        float g3 = G[g3Base + idx];
        const float sa = fabsf(g1) + fabsf(g2) + fabsf(g3);
        if (sa >= 1.0f) { g1 /= sa; g2 /= sa; g3 /= sa; }
        const float hu = hcol[t];
        const float hh = hcol[t + 1];
        const float hd = hcol[t + 2];
        const float hn = (1.0f - (g1 + g2 + g3)) * xt + g1 * hu + g2 * hh + g3 * hd;
        __syncthreads();
        hcol[t + 1] = hn;
        out[xBase + idx] = hn;
        __syncthreads();
    }
}

__global__ void max4_kernel(const float* __restrict__ a, const float* __restrict__ b,
                            const float* __restrict__ c, const float* __restrict__ d,
                            float* __restrict__ out, int n)
{
    const int i = blockIdx.x * blockDim.x + threadIdx.x;
    if (i >= n) return;
    out[i] = fmaxf(fmaxf(a[i], b[i]), fmaxf(c[i], d[i]));
}

__global__ void logsoftmax_kernel(const float* __restrict__ in, float* __restrict__ out,
                                  int Bn, int C, int HW)
{
    const int i = blockIdx.x * blockDim.x + threadIdx.x;
    const int total = Bn * HW;
    if (i >= total) return;
    const int b = i / HW;
    const int p = i - b * HW;
    const float* src = in + (size_t)b * C * HW + p;
    float m = -INFINITY;
    for (int c = 0; c < C; ++c) m = fmaxf(m, src[(size_t)c * HW]);
    float s = 0.0f;
    for (int c = 0; c < C; ++c) s += __expf(src[(size_t)c * HW] - m);
    const float lse = m + __logf(s);
    float* dst = out + (size_t)b * C * HW + p;
    for (int c = 0; c < C; ++c) dst[(size_t)c * HW] = src[(size_t)c * HW] - lse;
}

// ---------------------------------------------------------------------------
// Host-side helpers
// ---------------------------------------------------------------------------
struct ConvScratch { _Float16* col; _Float16* wf; };

static inline void conv_layer(hipStream_t s, ConvScratch cs,
                              const float* in, const float* w, const float* bias,
                              const float* res, float* out,
                              int Bn, int Cin, int Hin, int Win,
                              int Cout, int K, int stride, int pad, int act)
{
    const int Hout = (Hin + 2 * pad - K) / stride + 1;
    const int Wout = (Win + 2 * pad - K) / stride + 1;
    const int M    = Bn * Hout * Wout;
    const int KK   = Cin * K * K;
    const int Kpad = (KK + 31) & ~31;
    const int Npad = (Cout + (BN - 1)) & ~(BN - 1);

    im2col_f16_kernel<<<dim3(M, (Kpad + 255) / 256), 256, 0, s>>>(
        in, cs.col, Cin, Hin, Win, Hout, Wout, K, stride, pad, KK, Kpad);
    const int wtot = Npad * Kpad;
    wxform_f16_kernel<<<(wtot + 255) / 256, 256, 0, s>>>(w, cs.wf, Cout, KK, Kpad, wtot);
    gemm_wmma_kernel<<<dim3((M + BM - 1) / BM, Npad / BN), 128, 0, s>>>(
        cs.col, cs.wf, bias, res, out, M, Cout, Kpad, Hout, Wout, act);
}

static inline void launch_pool(hipStream_t s, const float* in, float* out,
                               int BC, int Hin, int Win, int relu_in)
{
    const int Hout = (Hin - 1) / 2 + 1;
    const int Wout = (Win - 1) / 2 + 1;
    const int n = BC * Hout * Wout;
    maxpool_kernel<<<(n + 255) / 256, 256, 0, s>>>(in, out, BC, Hin, Win, Hout, Wout, relu_in);
}

static inline void launch_up2(hipStream_t s, const float* in, const float* res,
                              float* out, int BC, int Hin, int Win)
{
    const int n = BC * 4 * Hin * Win;
    up2_kernel<<<(n + 255) / 256, 256, 0, s>>>(in, res, out, BC, Hin, Win);
}

extern "C" void kernel_launch(void* const* d_in, const int* in_sizes, int n_in,
                              void* d_out, int out_size, void* d_ws, size_t ws_size,
                              hipStream_t stream)
{
    (void)in_sizes; (void)n_in; (void)out_size; (void)ws_size;

    const int Bn = 2, NC = 21;
    const float* x   = (const float*)d_in[0];
    const float* rgb = (const float*)d_in[1];
    auto P = [&](int i) { return (const float*)d_in[i]; };
    // params (w,b pairs, _CFG order): 2 conv1_1 4 conv1_2 6 conv2_1 8 conv2_2
    // 10 conv3_1 12 conv3_2 14 conv3_3 16 conv4_1 18 conv4_2 20 conv4_3
    // 22 conv5_1 24 conv5_2 26 conv5_3 28 d_layer0 30 d_layer2 32 d_layer6
    // 34 d_layer8 36 mask_conv 38 post1 40 post2

    // ---- workspace carve-out (float units) ----
    float* base = (float*)d_ws;
    size_t off = 0;
    auto alloc = [&](size_t n) { float* p = base + off; off += (n + 7) & ~(size_t)7; return p; };
    float* buf256a = alloc(2u * 64 * 256 * 256);
    float* buf256b = alloc(2u * 64 * 256 * 256);
    float* p1      = alloc(2u * 64 * 128 * 128);
    float* c2a     = alloc(2u * 128 * 128 * 128);
    float* c2b     = alloc(2u * 128 * 128 * 128);
    float* p2      = alloc(2u * 128 * 64 * 64);
    float* c3a     = alloc(2u * 256 * 64 * 64);
    float* c3b     = alloc(2u * 256 * 64 * 64);
    float* conv3_3 = alloc(2u * 256 * 64 * 64);
    float* p3      = alloc(2u * 256 * 32 * 32);
    float* c4a     = alloc(2u * 512 * 32 * 32);
    float* c4b     = alloc(2u * 512 * 32 * 32);
    float* conv4_3 = alloc(2u * 512 * 32 * 32);
    float* p4      = alloc(2u * 512 * 16 * 16);
    float* c5a     = alloc(2u * 512 * 16 * 16);
    float* c5b     = alloc(2u * 512 * 16 * 16);
    float* conv5_3 = alloc(2u * 512 * 16 * 16);
    float* p5      = alloc(2u * 512 * 8 * 8);
    float* d0      = alloc(2u * 512 * 8 * 8);
    float* upd0    = alloc(2u * 512 * 16 * 16);
    float* d2res   = alloc(2u * 512 * 16 * 16);
    float* d3      = alloc(2u * 512 * 16 * 16);
    float* d4res   = alloc(2u * 512 * 32 * 32);
    float* upd4    = alloc(2u * 512 * 64 * 64);
    float* d6res   = alloc(2u * 256 * 64 * 64);
    float* upd6    = alloc(2u * 256 * 128 * 128);
    float* guide   = alloc(2u * 384 * 128 * 128);
    float* Xm      = alloc(2u * 32 * 128 * 128);
    float* o0      = alloc(2u * 32 * 128 * 128);
    float* o1      = alloc(2u * 32 * 128 * 128);
    float* o2      = alloc(2u * 32 * 128 * 128);
    float* o3      = alloc(2u * 32 * 128 * 128);
    float* mfuse   = alloc(2u * 32 * 128 * 128);
    // conv scratch: im2col max = 131072 x 576 f16 ; wf16 max = 512 x 4608 f16
    ConvScratch cs;
    cs.col = (_Float16*)alloc((size_t)131072 * 576 / 2 + 8);
    cs.wf  = (_Float16*)alloc((size_t)512 * 4608 / 2 + 8);

    // ---- VGG encoder ----
    conv_layer(stream, cs, rgb,     P(2),  P(3),  nullptr, buf256a, Bn,   3, 256, 256,  64, 3, 1, 1, ACT_RELU);
    conv_layer(stream, cs, buf256a, P(4),  P(5),  nullptr, buf256b, Bn,  64, 256, 256,  64, 3, 1, 1, ACT_NONE);
    launch_pool(stream, buf256b, p1, Bn * 64, 256, 256, 1);
    conv_layer(stream, cs, p1,      P(6),  P(7),  nullptr, c2a,     Bn,  64, 128, 128, 128, 3, 1, 1, ACT_RELU);
    conv_layer(stream, cs, c2a,     P(8),  P(9),  nullptr, c2b,     Bn, 128, 128, 128, 128, 3, 1, 1, ACT_NONE);
    launch_pool(stream, c2b, p2, Bn * 128, 128, 128, 1);
    conv_layer(stream, cs, p2,      P(10), P(11), nullptr, c3a,     Bn, 128,  64,  64, 256, 3, 1, 1, ACT_RELU);
    conv_layer(stream, cs, c3a,     P(12), P(13), nullptr, c3b,     Bn, 256,  64,  64, 256, 3, 1, 1, ACT_RELU);
    conv_layer(stream, cs, c3b,     P(14), P(15), nullptr, conv3_3, Bn, 256,  64,  64, 256, 3, 1, 1, ACT_NONE);
    launch_pool(stream, conv3_3, p3, Bn * 256, 64, 64, 1);
    conv_layer(stream, cs, p3,      P(16), P(17), nullptr, c4a,     Bn, 256,  32,  32, 512, 3, 1, 1, ACT_RELU);
    conv_layer(stream, cs, c4a,     P(18), P(19), nullptr, c4b,     Bn, 512,  32,  32, 512, 3, 1, 1, ACT_RELU);
    conv_layer(stream, cs, c4b,     P(20), P(21), nullptr, conv4_3, Bn, 512,  32,  32, 512, 3, 1, 1, ACT_NONE);
    launch_pool(stream, conv4_3, p4, Bn * 512, 32, 32, 1);
    conv_layer(stream, cs, p4,      P(22), P(23), nullptr, c5a,     Bn, 512,  16,  16, 512, 3, 1, 1, ACT_RELU);
    conv_layer(stream, cs, c5a,     P(24), P(25), nullptr, c5b,     Bn, 512,  16,  16, 512, 3, 1, 1, ACT_RELU);
    conv_layer(stream, cs, c5b,     P(26), P(27), nullptr, conv5_3, Bn, 512,  16,  16, 512, 3, 1, 1, ACT_NONE);
    launch_pool(stream, conv5_3, p5, Bn * 512, 16, 16, 1);

    // ---- decoder ----
    conv_layer(stream, cs, p5,   P(28), P(29), nullptr, d0,    Bn, 512,  8,  8, 512, 1, 1, 0, ACT_NONE);
    launch_up2(stream, d0, nullptr, upd0, Bn * 512, 8, 8);
    conv_layer(stream, cs, upd0, P(30), P(31), conv5_3, d2res, Bn, 512, 16, 16, 512, 3, 1, 1, ACT_RELU);
    conv_layer(stream, cs, d2res,P(30), P(31), nullptr, d3,    Bn, 512, 16, 16, 512, 3, 1, 1, ACT_RELU);
    launch_up2(stream, d3, conv4_3, d4res, Bn * 512, 16, 16);
    launch_up2(stream, d4res, nullptr, upd4, Bn * 512, 32, 32);
    conv_layer(stream, cs, upd4, P(32), P(33), conv3_3, d6res, Bn, 512, 64, 64, 256, 3, 1, 1, ACT_ELU);
    launch_up2(stream, d6res, nullptr, upd6, Bn * 256, 64, 64);
    conv_layer(stream, cs, upd6, P(34), P(35), nullptr, guide, Bn, 256, 128, 128, 384, 3, 1, 1, ACT_NONE);

    // ---- mask conv (k=4, stride 2) ----
    conv_layer(stream, cs, x, P(36), P(37), nullptr, Xm, Bn, NC, 256, 256, 32, 4, 2, 1, ACT_NONE);

    // ---- SPN: 4 directional scans, fuse, repeat, fuse ----
    const int C = 32, GC = 384, Hs = 128, Ws = 128;
    const int nSPN = Bn * C * Hs * Ws;
    dim3 sgrid(Bn * C);
    spn_scan_kernel<<<sgrid, 128, 0, stream>>>(Xm, guide, o0, C, GC,   0,  32,  64, Hs, Ws, 1, 0);
    spn_scan_kernel<<<sgrid, 128, 0, stream>>>(Xm, guide, o1, C, GC,  96, 128, 160, Hs, Ws, 1, 1);
    spn_scan_kernel<<<sgrid, 128, 0, stream>>>(Xm, guide, o2, C, GC, 192, 224, 256, Hs, Ws, 0, 0);
    spn_scan_kernel<<<sgrid, 128, 0, stream>>>(Xm, guide, o3, C, GC, 288, 320, 352, Hs, Ws, 0, 1);
    max4_kernel<<<(nSPN + 255) / 256, 256, 0, stream>>>(o0, o1, o2, o3, mfuse, nSPN);
    spn_scan_kernel<<<sgrid, 128, 0, stream>>>(mfuse, guide, o0, C, GC,   0,  32,  64, Hs, Ws, 1, 0);
    spn_scan_kernel<<<sgrid, 128, 0, stream>>>(mfuse, guide, o1, C, GC,  96, 128, 160, Hs, Ws, 1, 1);
    spn_scan_kernel<<<sgrid, 128, 0, stream>>>(mfuse, guide, o2, C, GC, 192, 224, 256, Hs, Ws, 0, 0);
    spn_scan_kernel<<<sgrid, 128, 0, stream>>>(mfuse, guide, o3, C, GC, 288, 320, 352, Hs, Ws, 0, 1);
    max4_kernel<<<(nSPN + 255) / 256, 256, 0, stream>>>(o0, o1, o2, o3, Xm, nSPN);

    // ---- post: up2 -> conv relu -> conv -> log_softmax ----
    float* upout  = c2a;      // 2*32*256*256 fits
    float* post1  = buf256a;  // 2*64*256*256
    float* logits = buf256b;  // 2*21*256*256 fits
    launch_up2(stream, Xm, nullptr, upout, Bn * 32, 128, 128);
    conv_layer(stream, cs, upout, P(38), P(39), nullptr, post1,  Bn, 32, 256, 256, 64, 3, 1, 1, ACT_RELU);
    conv_layer(stream, cs, post1, P(40), P(41), nullptr, logits, Bn, 64, 256, 256, NC, 3, 1, 1, ACT_NONE);

    const int npix = Bn * 256 * 256;
    logsoftmax_kernel<<<(npix + 255) / 256, 256, 0, stream>>>(logits, (float*)d_out, Bn, NC, 256 * 256);
}
